// CPM_59828894433933
// MI455X (gfx1250) — matrix operand order repaired and checked by reading, not compiled, not run on hardware
//
#include <hip/hip_runtime.h>
#include <hip/hip_bf16.h>

typedef __attribute__((ext_vector_type(16))) _Float16 v16h;
typedef __attribute__((ext_vector_type(8)))  _Float16 v8h;
typedef __attribute__((ext_vector_type(8)))  float    v8f;

#define B_DIM   64
#define C_DIM   2208
#define N_SP    196
#define K_CLS   200
#define M_TOT   (B_DIM * N_SP)          // 12544 rows, 784 tiles of 16
#define KT_N    16                      // k-tiles padded 13 -> 16 (uniform work per wave)
#define KT_REAL 13
#define KPAD    (KT_REAL * 16)          // 208 logit columns kept
#define CC_N    (C_DIM / 32)            // 69 c-chunks of 32
#define LDA     1128                    // LDS A row stride in halves (padded: conflict-free)
#define CN      (C_DIM * N_SP)          // 432768
#define OUT_ELEMS (B_DIM * CN)          // 27697152
#define TOTAL4  (OUT_ELEMS / 4)         // 6924288

// ---------------------------------------------------------------------------
// Kernel 1: repack fc_w (f32 [K,C]) into f16 B-fragment layout:
//   wpack[((kt*CC + cc)*32 + lane)*16 + i] = w[kt*16 + (lane&15)][cc*32 + (lane>=16)*8 + (i&7) + (i>>3)*16]
// (B 32x16 f16 layout: lane holds column n = lane&15; lanes 0-15 K={0..7,16..23}, lanes 16-31 +8)
// ---------------------------------------------------------------------------
__global__ void __launch_bounds__(256)
prepack_kernel(const float* __restrict__ fcw, _Float16* __restrict__ wpack) {
    const int t = blockIdx.x * 256 + threadIdx.x;    // total = 16*69*512 = 565248
    if (t >= KT_N * CC_N * 512) return;
    const int i    = t & 15;
    const int lane = (t >> 4) & 31;
    const int cc   = (t >> 9) % CC_N;
    const int kt   = t / (CC_N * 512);
    const int k = kt * 16 + (lane & 15);
    const int c = cc * 32 + ((lane >> 4) << 3) + (i & 7) + ((i >> 3) << 4);
    const float v = (k < K_CLS) ? fcw[k * C_DIM + c] : 0.0f;
    wpack[t] = (_Float16)v;
}

// ---------------------------------------------------------------------------
// Kernel 2: per 16-row tile of h=[B*N, C]: logits = h*W^T + b (WMMA f16->f32),
// then per-row softmax max value (1/sum(exp(l-lmax))) and argmax.
// ---------------------------------------------------------------------------
__global__ void __launch_bounds__(128)
gemm_softmax_kernel(const float* __restrict__ x, const float* __restrict__ fcb,
                    const _Float16* __restrict__ wpack,
                    float* __restrict__ maxval, int* __restrict__ maxids) {
    __shared__ __align__(16) _Float16 lds_a[16 * LDA];   // 36096 B (half of C at a time)
    __shared__ __align__(16) float    lgt[16 * KPAD];    // 13312 B
    __shared__ int   rowoff[16];
    __shared__ float pmax[128];
    __shared__ int   pidx[128];
    __shared__ float smx[16];
    __shared__ float psum[128];

    const int tid = threadIdx.x;
    const int m0  = blockIdx.x * 16;

    if (tid < 16) {
        const int m = m0 + tid;
        rowoff[tid] = (m / N_SP) * CN + (m % N_SP);      // x flat offset at c=0
    }

    const int wave  = tid >> 5;
    const int lane  = tid & 31;
    const int lhalf = lane >> 4;        // 0 or 1
    const int l15   = lane & 15;
    const int khalf = lhalf << 3;       // A-layout K offset: 0 or 8

    v8f acc[4] = {};
    const _Float16* arow = lds_a + l15 * LDA + khalf;

    // Stage C in two halves (1120 + 1088 cols) to keep static LDS < 64KB.
    for (int pass = 0; pass < 2; ++pass) {
        const int cbase = pass * 1120;
        const int cw    = pass ? 1088 : 1120;
        __syncthreads();                                 // prior pass fully consumed
        for (int idx = tid; idx < 16 * cw; idx += 128) {
            const int r = idx & 15;
            const int c = idx >> 4;
            lds_a[r * LDA + c] = (_Float16)x[rowoff[r] + (cbase + c) * N_SP];
        }
        __syncthreads();

        const int ccn = pass ? 34 : 35;
        for (int cc = 0; cc < ccn; ++cc) {
            const int gcc = pass * 35 + cc;              // global c-chunk index
            // A fragment (16-bit A 16x32 layout): lanes 0-15 K={0..7,16..23}, lanes 16-31 +8
            const _Float16* ap = arow + cc * 32;
            const v8h alo = *(const v8h*)(ap);
            const v8h ahi = *(const v8h*)(ap + 16);
            const v16h afrag = __builtin_shufflevector(alo, ahi,
                0, 1, 2, 3, 4, 5, 6, 7, 8, 9, 10, 11, 12, 13, 14, 15);
#pragma unroll
            for (int j = 0; j < 4; ++j) {                // uniform: every wave does 4 k-tiles
                const int kt = wave + (j << 2);
                const v16h bfrag =
                    *(const v16h*)(wpack + (((kt * CC_N + gcc) * 32 + lane) << 4));
                acc[j] = __builtin_amdgcn_wmma_f32_16x16x32_f16(
                    false, afrag, false, bfrag, (short)0, acc[j], false, false);
            }
        }
    }

    // Bias add + write logits to LDS; pad columns (k >= 200) -> -inf.
#pragma unroll
    for (int j = 0; j < 4; ++j) {
        const int kt = wave + (j << 2);
        if (kt < KT_REAL) {
            const int k = kt * 16 + l15;                 // D layout: col = lane&15
            const float bias = (k < K_CLS) ? fcb[k] : 0.0f;
#pragma unroll
            for (int r = 0; r < 8; ++r) {
                const int m = r + (lhalf << 3);          // D layout: row = r + 8*(lane>=16)
                lgt[m * KPAD + k] =
                    (k < K_CLS) ? (acc[j][r] + bias) : -__builtin_inff();
            }
        }
    }
    __syncthreads();

    // Row reductions: 8 threads per row, 26 columns each (8*26 == 208).
    {
        const int row = tid & 15;
        const int sl  = tid >> 4;
        const float* lr = lgt + row * KPAD;
        float mx = -__builtin_inff();
        int   mi = 0;
        const int c0 = sl * 26;
        for (int c = c0; c < c0 + 26; ++c) {
            const float v = lr[c];
            if (v > mx) { mx = v; mi = c; }
        }
        pmax[tid] = mx; pidx[tid] = mi;
    }
    __syncthreads();
    if (tid < 16) {
        float mx = -__builtin_inff();
        int   mi = 0;
        for (int s = 0; s < 8; ++s) {                    // ascending -> first occurrence
            const float v = pmax[s * 16 + tid];
            if (v > mx) { mx = v; mi = pidx[s * 16 + tid]; }
        }
        smx[tid] = mx;
        maxids[m0 + tid] = mi;
    }
    __syncthreads();
    {
        const int row = tid & 15;
        const int sl  = tid >> 4;
        const float* lr = lgt + row * KPAD;
        const float mx = smx[row];
        float s = 0.0f;
        const int c0 = sl * 26;
        for (int c = c0; c < c0 + 26; ++c) s += expf(lr[c] - mx);  // exp(-inf)=0 for pads
        psum[tid] = s;
    }
    __syncthreads();
    if (tid < 16) {
        float s = 0.0f;
        for (int sl = 0; sl < 8; ++sl) s += psum[sl * 16 + tid];
        maxval[m0 + tid] = 1.0f / s;                     // max of softmax row
    }
}

// ---------------------------------------------------------------------------
// Kernel 3: per-batch L2 normalize over N, deterministic histogram + L1 norm.
// ---------------------------------------------------------------------------
__global__ void __launch_bounds__(256)
norm_hist_kernel(const float* __restrict__ maxval, const int* __restrict__ maxids,
                 const int* __restrict__ flag,
                 float* __restrict__ normv, float* __restrict__ pr) {
    const int b = blockIdx.x;
    const int t = threadIdx.x;
    __shared__ float red[256];
    __shared__ float nv[N_SP];
    __shared__ int   sid[N_SP];

    float v = 0.0f;
    if (t < N_SP) {
        v = maxval[b * N_SP + t];
        sid[t] = maxids[b * N_SP + t];
    }
    red[t] = v * v;
    __syncthreads();
    for (int s = 128; s > 0; s >>= 1) {
        if (t < s) red[t] += red[t + s];
        __syncthreads();
    }
    const float l2 = sqrtf(red[0]);
    __syncthreads();
    red[t] = v;
    __syncthreads();
    for (int s = 128; s > 0; s >>= 1) {
        if (t < s) red[t] += red[t + s];
        __syncthreads();
    }
    const float sumv = red[0];

    const float rinv = 1.0f / fmaxf(l2, 1e-12f);
    const float nn = v * rinv;
    if (t < N_SP) {
        nv[t] = nn;
        normv[b * N_SP + t] = nn;
    }
    const float denom = fmaxf(sumv * rinv, 1e-12f);      // sum|p_r| == sum of norms
    const int fl = flag[0];
    __syncthreads();
    if (t < K_CLS) {
        float h = 0.0f;
        if (fl != 0) {
            for (int n = 0; n < N_SP; ++n) h += (sid[n] == t) ? nv[n] : 0.0f;
            h /= denom;
        }
        pr[b * K_CLS + t] = h;
    }
}

// ---------------------------------------------------------------------------
// Kernel 4: out[b,c,n] = x[b,c,n] * (1 + norm[b,n]), float4 vectorized.
// ---------------------------------------------------------------------------
__global__ void __launch_bounds__(256)
scale_residual_kernel(const float* __restrict__ x, const float* __restrict__ normv,
                      float* __restrict__ out) {
    const int i = blockIdx.x * 256 + threadIdx.x;
    if (i >= TOTAL4) return;
    const int e = i << 2;
    const int b = e / CN;
    const int n = e % N_SP;                              // 196 % 4 == 0 -> stays in-row
    const float4 xv = ((const float4*)x)[i];
    const float4 nr = *(const float4*)(normv + b * N_SP + n);
    float4 o;
    o.x = xv.x * (1.0f + nr.x);
    o.y = xv.y * (1.0f + nr.y);
    o.z = xv.z * (1.0f + nr.z);
    o.w = xv.w * (1.0f + nr.w);
    ((float4*)out)[i] = o;
}

// ---------------------------------------------------------------------------
extern "C" void kernel_launch(void* const* d_in, const int* in_sizes, int n_in,
                              void* d_out, int out_size, void* d_ws, size_t ws_size,
                              hipStream_t stream) {
    const float* x    = (const float*)d_in[0];
    const float* fcw  = (const float*)d_in[1];
    const float* fcb  = (const float*)d_in[2];
    const int*   flag = (const int*)d_in[3];
    float* outp = (float*)d_out;

    char* ws = (char*)d_ws;
    _Float16* wpack  = (_Float16*)(ws);                  // 1,130,496 B
    float*    maxval = (float*)(ws + 1130496);           //    50,176 B
    int*      maxids = (int*)(ws + 1180672);             //    50,176 B
    float*    normv  = (float*)(ws + 1230848);           //    50,176 B

    prepack_kernel<<<(KT_N * CC_N * 512) / 256, 256, 0, stream>>>(fcw, wpack);
    gemm_softmax_kernel<<<M_TOT / 16, 128, 0, stream>>>(x, fcb, wpack, maxval, maxids);
    norm_hist_kernel<<<B_DIM, 256, 0, stream>>>(maxval, maxids, flag, normv,
                                                outp + OUT_ELEMS);
    scale_residual_kernel<<<(TOTAL4 + 255) / 256, 256, 0, stream>>>(x, normv, outp);
}
